// _3DLPFEM_23192823398811
// MI455X (gfx1250) — matrix operand order repaired and checked by reading, not compiled, not run on hardware
//
#include <hip/hip_runtime.h>
#include <hip/hip_bf16.h>
#include <math.h>

// ---------------- dims ----------------
#define B_    4
#define C_    32
#define H_    128
#define W_    128
#define HW    16384          // H_*W_
#define NX    2097152        // B_*C_*HW
#define EMB   96
#define DI    192
#define NS    16             // SSM state dim N
#define RR    6              // dt rank
#define NB    8
#define LTOK  8192
#define MTOK  32768          // B_*LTOK
#define XDPAD 48             // padded stride for xd (R+2N = 38 -> 48)
#define NCH   (B_*DI)        // 768 scan channels
#define CLEN  256            // scan chunk length
#define NCK   (LTOK/CLEN)    // 32 chunks per sequence

typedef _Float16 v16h __attribute__((ext_vector_type(16)));
typedef _Float16 v8h  __attribute__((ext_vector_type(8)));
typedef float    v8f  __attribute__((ext_vector_type(8)));
typedef _Float16 half_t;

__device__ __forceinline__ float sigf(float x)  { return 1.f / (1.f + __expf(-x)); }
__device__ __forceinline__ float siluf(float x) { return x * sigf(x); }
__device__ __forceinline__ float lreluf(float x){ return x >= 0.f ? x : 0.2f * x; }
__device__ __forceinline__ float softplusf(float x) {
  return (x > 30.f) ? x : log1pf(__expf(x));
}

// f32 -> f16 converter with zero padding (weights padded to N multiple of 16*NSUB)
__global__ void w_to_f16_pad_k(const float* __restrict__ in, half_t* __restrict__ o,
                               int nreal, int ntot)
{
  const int g = blockIdx.x * blockDim.x + threadIdx.x;
  if (g < ntot) o[g] = (g < nreal) ? (half_t)in[g] : (half_t)0.f;
}

__global__ void f32_to_f16_k(const float* __restrict__ in, half_t* __restrict__ o, int n)
{
  const int g = blockIdx.x * blockDim.x + threadIdx.x;
  if (g < n) o[g] = (half_t)in[g];
}

// =====================================================================
// WMMA GEMM:  C[M,N](f32) = A[M,K](f16) @ W[N,K](f16)^T  (+bias, +=opt)
// Branch-free inner loop: one 16(M) x 16*NSUB(N) strip per wave, one A
// fragment feeds NSUB v_wmma_f32_16x16x32_f16 per 32-wide K step.
// W is pre-padded so 16*NSUB divides N exactly -> no masking anywhere.
// Wave id via readfirstlane => scalar tile guards, EXEC all-1s at WMMA,
// accumulators stay register-resident (no phi copies / spills).
// K mult of 32, M mult of 16; lda/ldw mult of 8 (16B-aligned rows).
// =====================================================================
template<int NSUB>
__global__ void gemm_xwT_wmma(const half_t* __restrict__ A, int lda,
                              const half_t* __restrict__ W,     // [Npad, Kd] row-major
                              const float* __restrict__ bias,   // nullable (only unpadded)
                              float* __restrict__ C, int ldc,
                              int mtiles, int nstrips, int Kd,
                              int accumulate)
{
  const int wave = __builtin_amdgcn_readfirstlane((int)(threadIdx.x >> 5));
  const int lane = threadIdx.x & 31;
  const int tile = blockIdx.x * (blockDim.x >> 5) + wave;   // scalar
  if (tile >= mtiles * nstrips) return;                     // scalar branch
  const int tm = tile % mtiles;                             // M-tile fastest
  const int tn = tile / mtiles;
  const int row0 = tm << 4;
  const int col0 = tn * (NSUB << 4);
  const int hi = lane >> 4;
  const int ln = lane & 15;

  const half_t* Arow = A + (size_t)(row0 + ln) * lda + (hi << 3);
  const half_t* Wr0  = W + (size_t)(col0 + ln) * Kd + (hi << 3);

  v8f acc[NSUB];
  #pragma unroll
  for (int s = 0; s < NSUB; ++s) acc[s] = (v8f){0.f,0.f,0.f,0.f,0.f,0.f,0.f,0.f};

  for (int k0 = 0; k0 < Kd; k0 += 32) {
    // A fragment (16x32 f16): lanes 0-15 K={0..7,16..23}, lanes 16-31 +8
    __builtin_prefetch(Arow + k0 + 32, 0, 1);
    const v8h a0 = *(const v8h*)(Arow + k0);
    const v8h a1 = *(const v8h*)(Arow + k0 + 16);
    v16h af;
    #pragma unroll
    for (int i = 0; i < 8; ++i) { af[i] = a0[i]; af[i + 8] = a1[i]; }

    #pragma unroll
    for (int s = 0; s < NSUB; ++s) {
      // B fragment (32x16 f16): n = lane&15, K={0..7,16..23} (+8 for hi)
      const half_t* Wp = Wr0 + (size_t)(s << 4) * Kd + k0;
      const v8h b0 = *(const v8h*)(Wp);
      const v8h b1 = *(const v8h*)(Wp + 16);
      v16h bf;
      #pragma unroll
      for (int i = 0; i < 8; ++i) { bf[i] = b0[i]; bf[i + 8] = b1[i]; }
      acc[s] = __builtin_amdgcn_wmma_f32_16x16x32_f16(
          false, af, false, bf, (short)0, acc[s], false, false);
    }
  }

  #pragma unroll
  for (int s = 0; s < NSUB; ++s) {
    const int n = col0 + (s << 4) + ln;
    const float bv = bias ? bias[n] : 0.f;
    #pragma unroll
    for (int r = 0; r < 8; ++r) {
      const size_t idx = (size_t)(row0 + r + (hi << 3)) * ldc + n;
      float v = acc[s][r] + bv;
      if (accumulate) v += C[idx];
      C[idx] = v;
    }
  }
}

// =====================================================================
// CBAM
// =====================================================================
__global__ void pool_hw_k(const float* __restrict__ x, float* __restrict__ avg,
                          float* __restrict__ mx)
{
  const int bc  = blockIdx.x;
  const int tid = threadIdx.x;
  const float* p = x + (size_t)bc * HW;
  float s = 0.f, m = -3.4e38f;
  for (int i = tid; i < HW; i += 256) { float v = p[i]; s += v; m = fmaxf(m, v); }
  __shared__ float ls[256], lm[256];
  ls[tid] = s; lm[tid] = m; __syncthreads();
  for (int st = 128; st; st >>= 1) {
    if (tid < st) { ls[tid] += ls[tid + st]; lm[tid] = fmaxf(lm[tid], lm[tid + st]); }
    __syncthreads();
  }
  if (tid == 0) { avg[bc] = ls[0] * (1.f / HW); mx[bc] = lm[0]; }
}

__global__ void cbam_ca_k(const float* __restrict__ avg, const float* __restrict__ mx,
                          const float* __restrict__ w1, const float* __restrict__ w2,
                          float* __restrict__ ca)
{
  const int bc = threadIdx.x;
  if (bc >= B_ * C_) return;
  const int b = bc >> 5, c = bc & 31;
  float o = 0.f;
  for (int src = 0; src < 2; ++src) {
    const float* v = (src == 0) ? (avg + b * C_) : (mx + b * C_);
    for (int j = 0; j < 2; ++j) {
      float h = 0.f;
      for (int cc = 0; cc < C_; ++cc) h += v[cc] * w1[j * C_ + cc];
      o += fmaxf(h, 0.f) * w2[c * 2 + j];
    }
  }
  ca[bc] = sigf(o);
}

__global__ void cbam_scale_stats_k(const float* __restrict__ x, const float* __restrict__ ca,
                                   float* __restrict__ xca, float* __restrict__ smap)
{
  const int g = blockIdx.x * blockDim.x + threadIdx.x;
  if (g >= B_ * HW) return;
  const int b = g / HW, hw = g % HW;
  float mean = 0.f, mxv = -3.4e38f;
  for (int c = 0; c < C_; ++c) {
    const size_t idx = (size_t)(b * C_ + c) * HW + hw;
    const float v = x[idx] * ca[b * C_ + c];
    xca[idx] = v;
    mean += v; mxv = fmaxf(mxv, v);
  }
  smap[(size_t)(b * 2 + 0) * HW + hw] = mean * (1.f / C_);
  smap[(size_t)(b * 2 + 1) * HW + hw] = mxv;
}

__global__ void cbam_spatial_k(const float* __restrict__ xca, const float* __restrict__ smap,
                               const float* __restrict__ saw, float* __restrict__ x1)
{
  const int g = blockIdx.x * blockDim.x + threadIdx.x;
  if (g >= B_ * HW) return;
  const int b = g / HW, hw = g % HW;
  const int h = hw >> 7, w = hw & 127;
  float s = 0.f;
  for (int ch = 0; ch < 2; ++ch)
    for (int dy = -3; dy <= 3; ++dy) {
      const int hh = h + dy; if (hh < 0 || hh >= H_) continue;
      for (int dx = -3; dx <= 3; ++dx) {
        const int ww = w + dx; if (ww < 0 || ww >= W_) continue;
        s += saw[ch * 49 + (dy + 3) * 7 + (dx + 3)] *
             smap[(size_t)(b * 2 + ch) * HW + hh * W_ + ww];
      }
    }
  const float sg = sigf(s);
  for (int c = 0; c < C_; ++c) {
    const size_t idx = (size_t)(b * C_ + c) * HW + hw;
    x1[idx] = xca[idx] * sg;
  }
}

// =====================================================================
// Patchify (f16 GEMM operand) / residual fuse
// =====================================================================
__global__ void patchify_k(const float* __restrict__ x, half_t* __restrict__ Ap)
{
  const int g = blockIdx.x * blockDim.x + threadIdx.x;
  if (g >= MTOK * 64) return;
  const int m = g >> 6, j = g & 63;
  const int b = m / LTOK, l = m % LTOK;
  const int cb = l >> 10, hb = (l >> 5) & 31, wb = l & 31;
  const int pc = j >> 4, ph = (j >> 2) & 3, pw = j & 3;
  Ap[g] = (half_t)x[(((size_t)(b * C_ + cb * 4 + pc) * H_) + hb * 4 + ph) * W_ + wb * 4 + pw];
}

__global__ void fuse_residual_k(const float* __restrict__ x, const float* __restrict__ x1,
                                const float* __restrict__ y64, float* __restrict__ xs)
{
  const int g = blockIdx.x * blockDim.x + threadIdx.x;
  if (g >= NX) return;
  const int w = g & 127, h = (g >> 7) & 127, c = (g >> 14) & 31, b = g >> 19;
  const int cb = c >> 2, pc = c & 3, hb = h >> 2, ph = h & 3, wb = w >> 2, pw = w & 3;
  const int l = (cb * 32 + hb) * 32 + wb;
  const size_t m = (size_t)b * LTOK + l;
  const int j = (pc * 4 + ph) * 4 + pw;
  xs[g] = x[g] + x1[g] + y64[m * 64 + j];
}

// =====================================================================
// Mamba block pieces
// =====================================================================
__global__ void layernorm_k(const float* __restrict__ t, const float* __restrict__ w,
                            const float* __restrict__ b, half_t* __restrict__ h)
{
  const int wave = __builtin_amdgcn_readfirstlane((int)(threadIdx.x >> 5));
  const int lane = threadIdx.x & 31;
  const int row = blockIdx.x * 8 + wave;                    // scalar
  if (row >= MTOK) return;
  const float* x = t + (size_t)row * EMB;
  const int e = lane * 3;
  float v0 = x[e], v1 = x[e + 1], v2 = x[e + 2];
  float s = v0 + v1 + v2;
  for (int o = 16; o; o >>= 1) s += __shfl_xor(s, o, 32);
  const float mu = s * (1.f / EMB);
  const float d0 = v0 - mu, d1 = v1 - mu, d2 = v2 - mu;
  float q = d0 * d0 + d1 * d1 + d2 * d2;
  for (int o = 16; o; o >>= 1) q += __shfl_xor(q, o, 32);
  const float inv = rsqrtf(q * (1.f / EMB) + 1e-5f);
  half_t* hp = h + (size_t)row * EMB;
  hp[e]     = (half_t)(d0 * inv * w[e]     + b[e]);
  hp[e + 1] = (half_t)(d1 * inv * w[e + 1] + b[e + 1]);
  hp[e + 2] = (half_t)(d2 * inv * w[e + 2] + b[e + 2]);
}

// causal depthwise conv + SiLU; emits f32 (scan input) and f16 (GEMM A)
__global__ void dwconv_silu_k(const float* __restrict__ xz, const float* __restrict__ cw,
                              const float* __restrict__ cb, float* __restrict__ xcD,
                              half_t* __restrict__ xcD16)
{
  const int g = blockIdx.x * blockDim.x + threadIdx.x;
  if (g >= MTOK * DI) return;
  const int d = g % DI;
  const int m = g / DI;
  const int l = m % LTOK;
  const float x0 = xz[(size_t)m * 384 + d];
  const float x1 = (l >= 1) ? xz[(size_t)(m - 1) * 384 + d] : 0.f;
  const float x2 = (l >= 2) ? xz[(size_t)(m - 2) * 384 + d] : 0.f;
  const float y = cw[d * 3 + 0] * x2 + cw[d * 3 + 1] * x1 + cw[d * 3 + 2] * x0 + cb[d];
  const float v = siluf(y);
  xcD[(size_t)m * DI + d] = v;
  xcD16[(size_t)m * DI + d] = (half_t)v;
}

// =====================================================================
// Chunked selective scan (3-pass associative scan).
// Half-wave (16 lanes = 16 states) owns one (channel, chunk).
// =====================================================================
__global__ void ssm_scan_part1(const float* __restrict__ u, const float* __restrict__ xd,
                               const float* __restrict__ dtw, const float* __restrict__ dtb,
                               const float* __restrict__ Alog,
                               float* __restrict__ chA, float* __restrict__ chB,
                               int reverse)
{
  const int lane = threadIdx.x & 31;
  const int wave = threadIdx.x >> 5;
  const int half = lane >> 4;
  const int n = lane & 15;
  const int unit = (blockIdx.x * (blockDim.x >> 5) + wave) * 2 + half;
  if (unit >= NCH * NCK) return;
  const int ch = unit >> 5;            // /NCK
  const int c  = unit & (NCK - 1);
  const int b = ch / DI, d = ch % DI;
  const float A    = -__expf(Alog[d * NS + n]);
  const float dtwn = (n < RR) ? dtw[d * RR + n] : 0.f;
  const float dtbd = dtb[d];
  const int base   = lane & 16;
  float h = 0.f, P = 1.f;
  const int t0 = c * CLEN;
  for (int t = t0; t < t0 + CLEN; ++t) {
    const int l = reverse ? (LTOK - 1 - t) : t;
    const size_t m = (size_t)b * LTOK + l;
    const float* xr = xd + m * XDPAD;
    float uval = (n == 0) ? u[m * DI + d] : 0.f;
    uval = __shfl(uval, base, 32);
    float dr = (n < RR) ? xr[n] * dtwn : 0.f;
    dr += __shfl_xor(dr, 8, 32);
    dr += __shfl_xor(dr, 4, 32);
    dr += __shfl_xor(dr, 2, 32);
    dr += __shfl_xor(dr, 1, 32);
    const float dt = softplusf(dr + dtbd);
    const float dA = __expf(dt * A);
    P *= dA;
    h = dA * h + (dt * uval) * xr[RR + n];
  }
  chA[(size_t)unit * NS + n] = P;
  chB[(size_t)unit * NS + n] = h;
}

__global__ void ssm_scan_part2(const float* __restrict__ chA, const float* __restrict__ chB,
                               float* __restrict__ hst)
{
  const int id = blockIdx.x * blockDim.x + threadIdx.x;
  if (id >= NCH * NS) return;
  const int ch = id / NS, n = id % NS;
  float h = 0.f;
  for (int c = 0; c < NCK; ++c) {
    const size_t o = ((size_t)ch * NCK + c) * NS + n;
    hst[o] = h;
    h = chA[o] * h + chB[o];
  }
}

__global__ void ssm_scan_part3(const float* __restrict__ u, const float* __restrict__ xd,
                               const float* __restrict__ dtw, const float* __restrict__ dtb,
                               const float* __restrict__ Alog, const float* __restrict__ Dp,
                               const float* __restrict__ mgp, const float* __restrict__ hst,
                               float* __restrict__ ys, int reverse, int accumulate)
{
  const int lane = threadIdx.x & 31;
  const int wave = threadIdx.x >> 5;
  const int half = lane >> 4;
  const int n = lane & 15;
  const int unit = (blockIdx.x * (blockDim.x >> 5) + wave) * 2 + half;
  if (unit >= NCH * NCK) return;
  const int ch = unit >> 5;
  const int c  = unit & (NCK - 1);
  const int b = ch / DI, d = ch % DI;
  const float A    = -__expf(Alog[d * NS + n]);
  const float dtwn = (n < RR) ? dtw[d * RR + n] : 0.f;
  const float dtbd = dtb[d];
  const float Dpd  = Dp[d];
  const float mg   = mgp[0];
  const int base   = lane & 16;
  float h = hst[(size_t)unit * NS + n];
  const int t0 = c * CLEN;
  for (int t = t0; t < t0 + CLEN; ++t) {
    const int l = reverse ? (LTOK - 1 - t) : t;
    const size_t m = (size_t)b * LTOK + l;
    const float* xr = xd + m * XDPAD;
    float uval = (n == 0) ? u[m * DI + d] : 0.f;
    uval = __shfl(uval, base, 32);
    float dr = (n < RR) ? xr[n] * dtwn : 0.f;
    dr += __shfl_xor(dr, 8, 32);
    dr += __shfl_xor(dr, 4, 32);
    dr += __shfl_xor(dr, 2, 32);
    dr += __shfl_xor(dr, 1, 32);
    const float dt = softplusf(dr + dtbd);
    h = __expf(dt * A) * h + (dt * uval) * xr[RR + n];
    float cc = h * xr[RR + NS + n];
    cc += __shfl_xor(cc, 8, 32);
    cc += __shfl_xor(cc, 4, 32);
    cc += __shfl_xor(cc, 2, 32);
    cc += __shfl_xor(cc, 1, 32);
    if (n == 0) {
      const float y = mg * (cc + Dpd * uval);
      float* yp = ys + m * DI + d;
      if (accumulate) *yp += y; else *yp = y;
    }
  }
}

// ysil16 = f16( ys * silu(z) ), z = xz cols 192..383
__global__ void gate_mul_k(const float* __restrict__ xz, const float* __restrict__ ys,
                           half_t* __restrict__ ysil16)
{
  const int g = blockIdx.x * blockDim.x + threadIdx.x;
  if (g >= MTOK * DI) return;
  const int d = g % DI;
  const size_t m = g / DI;
  const float z = xz[m * 384 + DI + d];
  ysil16[m * DI + d] = (half_t)(ys[m * DI + d] * siluf(z));
}

// =====================================================================
// BatchNorm (training-mode batch stats)
// =====================================================================
__global__ void bn_stats_k(const float* __restrict__ x, float* __restrict__ stats)
{
  const int c = blockIdx.x, tid = threadIdx.x;
  float s = 0.f, q = 0.f;
  for (int i = tid; i < B_ * HW; i += 256) {
    const int b = i / HW, hw = i % HW;
    const float v = x[(size_t)(b * C_ + c) * HW + hw];
    s += v; q += v * v;
  }
  __shared__ float ls[256], lq[256];
  ls[tid] = s; lq[tid] = q; __syncthreads();
  for (int st = 128; st; st >>= 1) {
    if (tid < st) { ls[tid] += ls[tid + st]; lq[tid] += lq[tid + st]; }
    __syncthreads();
  }
  if (tid == 0) {
    const float mu = ls[0] / (float)(B_ * HW);
    stats[c] = mu;
    stats[C_ + c] = lq[0] / (float)(B_ * HW) - mu * mu;
  }
}

__global__ void bn_apply_k(const float* __restrict__ x, const float* __restrict__ stats,
                           const float* __restrict__ w, const float* __restrict__ b,
                           float* __restrict__ y)
{
  const int g = blockIdx.x * blockDim.x + threadIdx.x;
  if (g >= NX) return;
  const int c = (g >> 14) & 31;
  y[g] = (x[g] - stats[c]) * rsqrtf(stats[C_ + c] + 1e-5f) * w[c] + b[c];
}

// =====================================================================
// SCPA convs
// =====================================================================
__global__ void conv1x1_k(const float* __restrict__ in, const float* __restrict__ w,
                          int IC, int OC, float* __restrict__ out, int act)
{
  const int g = blockIdx.x * blockDim.x + threadIdx.x;
  if (g >= B_ * OC * HW) return;
  const int hw = g % HW, oc = (g / HW) % OC, b = g / (HW * OC);
  float s = 0.f;
  for (int ic = 0; ic < IC; ++ic)
    s += w[oc * IC + ic] * in[(size_t)(b * IC + ic) * HW + hw];
  if (act == 1) s = lreluf(s);
  out[(size_t)(b * OC + oc) * HW + hw] = s;
}

__global__ void conv3x3_k(const float* __restrict__ in, const float* __restrict__ w,
                          float* __restrict__ out, int act)   // 16 -> 16, pad 1
{
  const int g = blockIdx.x * blockDim.x + threadIdx.x;
  if (g >= B_ * 16 * HW) return;
  const int hw = g % HW, oc = (g / HW) % 16, b = g / (HW * 16);
  const int h = hw >> 7, wx = hw & 127;
  float s = 0.f;
  for (int ic = 0; ic < 16; ++ic) {
    const float* ip = in + (size_t)(b * 16 + ic) * HW;
    const float* wp = w + (oc * 16 + ic) * 9;
    for (int dy = -1; dy <= 1; ++dy) {
      const int hh = h + dy; if (hh < 0 || hh >= H_) continue;
      for (int dx = -1; dx <= 1; ++dx) {
        const int ww = wx + dx; if (ww < 0 || ww >= W_) continue;
        s += wp[(dy + 1) * 3 + (dx + 1)] * ip[hh * W_ + ww];
      }
    }
  }
  if (act == 1) s = lreluf(s);
  out[(size_t)(b * 16 + oc) * HW + hw] = s;
}

__global__ void gate_comb_k(const float* __restrict__ t2, const float* __restrict__ t3,
                            float* __restrict__ b1)
{
  const int g = blockIdx.x * blockDim.x + threadIdx.x;
  if (g >= B_ * 16 * HW) return;
  b1[g] = lreluf(t3[g] * sigf(t2[g]));
}

__global__ void scpa_final_k(const float* __restrict__ a1, const float* __restrict__ b2,
                             const float* __restrict__ c3w, const float* __restrict__ bn,
                             const float* __restrict__ xs, float* __restrict__ out)
{
  const int g = blockIdx.x * blockDim.x + threadIdx.x;
  if (g >= NX) return;
  const int hw = g % HW, c = (g / HW) % C_, b = g / (HW * C_);
  float s = 0.f;
  for (int ic = 0; ic < C_; ++ic) {
    const float v = (ic < 16) ? a1[(size_t)(b * 16 + ic) * HW + hw]
                              : b2[(size_t)(b * 16 + ic - 16) * HW + hw];
    s += c3w[c * C_ + ic] * v;
  }
  out[g] = s + bn[g] + xs[g];
}

// =====================================================================
// Host launch
// =====================================================================
static inline int cdiv(long long a, long long b) { return (int)((a + b - 1) / b); }

extern "C" void kernel_launch(void* const* d_in, const int* in_sizes, int n_in,
                              void* d_out, int out_size, void* d_ws, size_t ws_size,
                              hipStream_t stream) {
  (void)in_sizes; (void)n_in; (void)out_size; (void)ws_size;

  const float* x      = (const float*)d_in[0];
  const float* ca_w1  = (const float*)d_in[1];
  const float* ca_w2  = (const float*)d_in[2];
  const float* sa_w   = (const float*)d_in[3];
  const float* pe_w   = (const float*)d_in[4];
  const float* pe_b   = (const float*)d_in[5];
  const float* pu_w   = (const float*)d_in[6];
  const float* pu_b   = (const float*)d_in[7];
  const float* ln_w   = (const float*)d_in[8];
  const float* ln_b   = (const float*)d_in[9];
  const float* in_w   = (const float*)d_in[10];
  const float* conv_w = (const float*)d_in[11];
  const float* conv_b = (const float*)d_in[12];
  const float* xp_w   = (const float*)d_in[13];
  const float* dt_w   = (const float*)d_in[14];
  const float* dt_b   = (const float*)d_in[15];
  const float* Alog   = (const float*)d_in[16];
  const float* Dp     = (const float*)d_in[17];
  const float* mg     = (const float*)d_in[18];
  const float* out_w  = (const float*)d_in[19];
  const float* bn_w   = (const float*)d_in[20];
  const float* bn_b   = (const float*)d_in[21];
  const float* c1a    = (const float*)d_in[22];
  const float* c1b    = (const float*)d_in[23];
  const float* k1     = (const float*)d_in[24];
  const float* k2     = (const float*)d_in[25];
  const float* k3     = (const float*)d_in[26];
  const float* k4     = (const float*)d_in[27];
  const float* c3     = (const float*)d_in[28];
  float* out = (float*)d_out;

  char* wsb = (char*)d_ws;
  size_t wof = 0;
  auto wsraw = [&](size_t bytes) -> void* {
    void* p = (void*)(wsb + wof);
    wof = (wof + bytes + 255) & ~(size_t)255;
    return p;
  };
  auto wsf = [&](size_t nfl) -> float*  { return (float*) wsraw(nfl * 4); };
  auto wsh = [&](size_t nh)  -> half_t* { return (half_t*)wsraw(nh * 2); };

  float*  pool_avg = wsf(B_ * C_);
  float*  pool_max = wsf(B_ * C_);
  float*  ca       = wsf(B_ * C_);
  float*  bnstats  = wsf(2 * C_);
  half_t* w16      = wsh(2 * DI * EMB);          // max padded weight = 384x96
  float*  smap     = wsf((size_t)B_ * 2 * HW);
  float*  chA      = wsf((size_t)NCH * NCK * NS);
  float*  chB      = wsf((size_t)NCH * NCK * NS);
  float*  hst      = wsf((size_t)NCH * NCK * NS);
  float*  xca      = wsf(NX);                    // reused as BN output later
  float*  x1       = wsf(NX);
  half_t* Ap16     = wsh((size_t)MTOK * 64);
  float*  y64      = wsf((size_t)MTOK * 64);
  float*  tbuf     = wsf((size_t)MTOK * EMB);
  half_t* tbuf16   = wsh((size_t)MTOK * EMB);    // doubles as hbuf16 (LN out)
  float*  xz       = wsf((size_t)MTOK * 384);    // reused for SCPA temps
  float*  xcD      = wsf((size_t)MTOK * DI);
  half_t* xcD16    = wsh((size_t)MTOK * DI);
  half_t* ysil16   = wsh((size_t)MTOK * DI);
  float*  xd0      = wsf((size_t)MTOK * XDPAD);
  float*  xd1      = wsf((size_t)MTOK * XDPAD);
  float*  ysb      = wsf((size_t)MTOK * DI);
  float*  xs       = wsf(NX);

  // GEMM dispatcher: pad W to Npad (mult of 16), pick NSUB dividing Npad/16
  auto gemm = [&](const half_t* A16, int lda, const float* Wf, const float* bias,
                  float* Cc, int ldc, int M, int Nreal, int Kd, int accum) {
    const int Npad = (Nreal + 15) & ~15;
    const int nsub16 = Npad >> 4;
    const int NSUB = (nsub16 % 4 == 0) ? 4 : (nsub16 % 3 == 0) ? 3
                   : (nsub16 % 2 == 0) ? 2 : 1;
    const int nstrips = nsub16 / NSUB;
    w_to_f16_pad_k<<<cdiv(Npad * Kd, 256), 256, 0, stream>>>(Wf, w16,
                                                             Nreal * Kd, Npad * Kd);
    const int mtiles = M >> 4;
    const int blocks = cdiv((long long)mtiles * nstrips, 4);
    switch (NSUB) {
      case 4: gemm_xwT_wmma<4><<<blocks, 128, 0, stream>>>(A16, lda, w16, bias, Cc, ldc,
                                                           mtiles, nstrips, Kd, accum); break;
      case 3: gemm_xwT_wmma<3><<<blocks, 128, 0, stream>>>(A16, lda, w16, bias, Cc, ldc,
                                                           mtiles, nstrips, Kd, accum); break;
      case 2: gemm_xwT_wmma<2><<<blocks, 128, 0, stream>>>(A16, lda, w16, bias, Cc, ldc,
                                                           mtiles, nstrips, Kd, accum); break;
      default: gemm_xwT_wmma<1><<<blocks, 128, 0, stream>>>(A16, lda, w16, bias, Cc, ldc,
                                                            mtiles, nstrips, Kd, accum); break;
    }
  };

  // ---------- CBAM ----------
  pool_hw_k<<<B_ * C_, 256, 0, stream>>>(x, pool_avg, pool_max);
  cbam_ca_k<<<1, 128, 0, stream>>>(pool_avg, pool_max, ca_w1, ca_w2, ca);
  cbam_scale_stats_k<<<cdiv(B_ * HW, 256), 256, 0, stream>>>(x, ca, xca, smap);
  cbam_spatial_k<<<cdiv(B_ * HW, 256), 256, 0, stream>>>(xca, smap, sa_w, x1);

  // ---------- patch embed ----------
  patchify_k<<<cdiv((long long)MTOK * 64, 256), 256, 0, stream>>>(x, Ap16);
  gemm(Ap16, 64, pe_w, pe_b, tbuf, EMB, MTOK, EMB, 64, 0);   // N=96 -> NSUB=3

  // ---------- Mamba blocks ----------
  const int scan_units  = NCH * NCK;                 // 24576 half-wave units
  const int scan_blocks = cdiv(scan_units, 16);
  for (int blk = 0; blk < NB; ++blk) {
    layernorm_k<<<MTOK / 8, 256, 0, stream>>>(tbuf, ln_w + blk * EMB, ln_b + blk * EMB, tbuf16);
    gemm(tbuf16, EMB, in_w + (size_t)blk * 2 * DI * EMB, nullptr, xz, 384,
         MTOK, 2 * DI, EMB, 0);                              // N=384 -> NSUB=4
    dwconv_silu_k<<<cdiv((long long)MTOK * DI, 256), 256, 0, stream>>>(
        xz, conv_w + blk * DI * 3, conv_b + blk * DI, xcD, xcD16);
    for (int k = 0; k < 2; ++k) {
      float* xdk = k ? xd1 : xd0;
      const size_t pk = (size_t)(blk * 2 + k);
      gemm(xcD16, DI, xp_w + pk * (RR + 2 * NS) * DI, nullptr, xdk, XDPAD,
           MTOK, RR + 2 * NS, DI, 0);                        // N=38 -> pad 48, NSUB=3
      ssm_scan_part1<<<scan_blocks, 256, 0, stream>>>(xcD, xdk,
          dt_w + pk * DI * RR, dt_b + pk * DI, Alog + pk * DI * NS, chA, chB, k);
      ssm_scan_part2<<<cdiv(NCH * NS, 256), 256, 0, stream>>>(chA, chB, hst);
      ssm_scan_part3<<<scan_blocks, 256, 0, stream>>>(xcD, xdk,
          dt_w + pk * DI * RR, dt_b + pk * DI, Alog + pk * DI * NS,
          Dp + pk * DI, mg + pk, hst, ysb, /*reverse=*/k, /*accumulate=*/k);
    }
    gate_mul_k<<<cdiv((long long)MTOK * DI, 256), 256, 0, stream>>>(xz, ysb, ysil16);
    gemm(ysil16, DI, out_w + (size_t)blk * EMB * DI, nullptr, tbuf, EMB,
         MTOK, EMB, DI, 1);                                  // N=96 -> NSUB=3
  }

  // ---------- patch unembed + residuals ----------
  f32_to_f16_k<<<cdiv((long long)MTOK * EMB, 256), 256, 0, stream>>>(tbuf, tbuf16, MTOK * EMB);
  gemm(tbuf16, EMB, pu_w, pu_b, y64, 64, MTOK, 64, EMB, 0);  // N=64 -> NSUB=4
  fuse_residual_k<<<cdiv(NX, 256), 256, 0, stream>>>(x, x1, y64, xs);

  // ---------- batchnorm ----------
  bn_stats_k<<<C_, 256, 0, stream>>>(xs, bnstats);
  bn_apply_k<<<cdiv(NX, 256), 256, 0, stream>>>(xs, bnstats, bn_w, bn_b, xca);

  // ---------- SCPA (temps inside xz region) ----------
  const size_t PLANE16 = (size_t)B_ * 16 * HW;
  float* a0 = xz + 0 * PLANE16;
  float* a1 = xz + 1 * PLANE16;
  float* b0 = xz + 2 * PLANE16;
  float* t2 = xz + 3 * PLANE16;
  float* t3 = xz + 4 * PLANE16;
  float* b2 = xz + 5 * PLANE16;
  const int g16 = cdiv(PLANE16, 256);
  conv1x1_k<<<g16, 256, 0, stream>>>(xca, c1a, C_, 16, a0, 1);
  conv3x3_k<<<g16, 256, 0, stream>>>(a0, k1, a1, 1);
  conv1x1_k<<<g16, 256, 0, stream>>>(xca, c1b, C_, 16, b0, 1);
  conv1x1_k<<<g16, 256, 0, stream>>>(b0, k2, 16, 16, t2, 0);
  conv3x3_k<<<g16, 256, 0, stream>>>(b0, k3, t3, 0);
  gate_comb_k<<<g16, 256, 0, stream>>>(t2, t3, b0);
  conv3x3_k<<<g16, 256, 0, stream>>>(b0, k4, b2, 1);
  scpa_final_k<<<cdiv(NX, 256), 256, 0, stream>>>(a1, b2, c3, xca, xs, out);
}
